// AR_Transcriber_47313359732944
// MI455X (gfx1250) — compile-verified, matrix operand order checked by reading, not run on hardware
//
#include <hip/hip_runtime.h>
#include <hip/hip_bf16.h>

// ---------------------------------------------------------------------------
// AR transcriber for MI455X (gfx1250, wave32, WMMA).
//  - conv frontend: direct f32 kernels (parallel, ~157 GFLOP total)
//  - FC + hoisted LSTM input projection: bf16 WMMA GEMM (v_wmma_f32_16x16x32_bf16)
//  - sequential AR LSTM: persistent 32-wave workgroup, state in LDS,
//    per-step matmuls via WMMA, bf16 weights resident in L2, and the
//    precomputed input-projection stream P[t] double-buffered in LDS via
//    GLOBAL_LOAD_ASYNC_TO_LDS_B128 (ASYNCcnt split-wait) to hide L2 latency.
// ---------------------------------------------------------------------------

typedef __attribute__((ext_vector_type(16))) __bf16        v16bf;
typedef __attribute__((ext_vector_type(8)))  float         v8f;
typedef __attribute__((ext_vector_type(4)))  unsigned int  u32x4;

union BfVec { v16bf v; u32x4 q[2]; };

__device__ __forceinline__ unsigned short f32_to_bf16(float f) {
    unsigned int u = __float_as_uint(f);
    unsigned int r = (u + 0x7FFFu + ((u >> 16) & 1u)) >> 16;   // RNE
    return (unsigned short)r;
}

__device__ __forceinline__ float sigf(float x) { return 1.0f / (1.0f + expf(-x)); }

// Load one WMMA 16-bit operand slice (A-style or B-style; both use the same
// per-lane K packing per CDNA5 ISA 7.12.2): lane<16 -> K {k..k+7, k+16..k+23},
// lane>=16 -> K {k+8..k+15, k+24..k+31}. rowoff = row * stride (elements).
__device__ __forceinline__ v16bf ld_bf16_op(const unsigned short* base, long rowoff,
                                            int k, int lane) {
    BfVec u;
    long o = rowoff + k + (long)((lane >> 4) << 3);
    u.q[0] = *reinterpret_cast<const u32x4*>(base + o);
    u.q[1] = *reinterpret_cast<const u32x4*>(base + o + 16);
    return u.v;
}

// ---------------------------------------------------------------------------
// problem constants
// ---------------------------------------------------------------------------
constexpr int BATCH = 4, T = 2048, F_IN = 229;
constexpr int C1 = 48, C2 = 96;
constexpr int F2 = 114, F3 = 57;           // after the two 1x2 pools
constexpr int DC = 768;                    // D_CONV == D_LSTM
constexpr int G4 = 3072;                   // 4 * 768 gates
constexpr int KFC = C2 * F3;               // 5472
constexpr int DIN0 = 944;                  // 768 + 176
constexpr int KPREV = 176, KPREVP = 192;   // prev embed dim (padded to mult of 32)
constexpr int NPOST = 440, NPOSTP = 448;   // 88*5 (padded to mult of 16)
constexpr int NPITCH = 88, NCLS = 5;
constexpr int PSTEP_BYTES = 4 * G4 * 4;    // one step of P: 4x3072 f32 = 49152 B

// ---------------------------------------------------------------------------
// conv frontend (direct f32)
// ---------------------------------------------------------------------------
__global__ __launch_bounds__(256) void conv1_kernel(
    const float* __restrict__ mel, const float* __restrict__ w,
    const float* __restrict__ bb, const float* __restrict__ s,
    const float* __restrict__ sb, float* __restrict__ out) {
    int idx = blockIdx.x * 256 + threadIdx.x;
    const int total = BATCH * C1 * T * F_IN;
    if (idx >= total) return;
    int f = idx % F_IN;
    int t = (idx / F_IN) % T;
    int c = (idx / (F_IN * T)) % C1;
    int b = idx / (F_IN * T * C1);
    float acc = 0.f;
    const float* wc = w + c * 9;
    for (int kh = 0; kh < 3; ++kh) {
        int t2 = t + kh - 1;
        if ((unsigned)t2 >= (unsigned)T) continue;
        for (int kw = 0; kw < 3; ++kw) {
            int f2 = f + kw - 1;
            if ((unsigned)f2 >= (unsigned)F_IN) continue;
            acc += mel[(b * T + t2) * F_IN + f2] * wc[kh * 3 + kw];
        }
    }
    float v = (acc + bb[c]) * s[c] + sb[c];
    out[idx] = v > 0.f ? v : 0.f;
}

__global__ __launch_bounds__(256) void conv2_kernel(
    const float* __restrict__ in, const float* __restrict__ w,
    const float* __restrict__ bb, const float* __restrict__ s,
    const float* __restrict__ sb, float* __restrict__ out) {
    int idx = blockIdx.x * 256 + threadIdx.x;
    const int total = BATCH * C1 * T * F_IN;
    if (idx >= total) return;
    int f = idx % F_IN;
    int t = (idx / F_IN) % T;
    int c = (idx / (F_IN * T)) % C1;
    int b = idx / (F_IN * T * C1);
    float acc = 0.f;
    const float* wc = w + c * C1 * 9;
    for (int ic = 0; ic < C1; ++ic) {
        const float* ip = in + ((b * C1 + ic) * T) * F_IN;
        const float* wp = wc + ic * 9;
        for (int kh = 0; kh < 3; ++kh) {
            int t2 = t + kh - 1;
            if ((unsigned)t2 >= (unsigned)T) continue;
            for (int kw = 0; kw < 3; ++kw) {
                int f2 = f + kw - 1;
                if ((unsigned)f2 >= (unsigned)F_IN) continue;
                acc += ip[t2 * F_IN + f2] * wp[kh * 3 + kw];
            }
        }
    }
    float v = (acc + bb[c]) * s[c] + sb[c];
    out[idx] = v > 0.f ? v : 0.f;
}

__global__ __launch_bounds__(256) void conv3_kernel(
    const float* __restrict__ in, const float* __restrict__ w,
    const float* __restrict__ bb, const float* __restrict__ s,
    const float* __restrict__ sb, float* __restrict__ out) {
    int idx = blockIdx.x * 256 + threadIdx.x;
    const int total = BATCH * C2 * T * F2;
    if (idx >= total) return;
    int f = idx % F2;
    int t = (idx / F2) % T;
    int c = (idx / (F2 * T)) % C2;
    int b = idx / (F2 * T * C2);
    float acc = 0.f;
    const float* wc = w + c * C1 * 9;
    for (int ic = 0; ic < C1; ++ic) {
        const float* ip = in + ((b * C1 + ic) * T) * F2;
        const float* wp = wc + ic * 9;
        for (int kh = 0; kh < 3; ++kh) {
            int t2 = t + kh - 1;
            if ((unsigned)t2 >= (unsigned)T) continue;
            for (int kw = 0; kw < 3; ++kw) {
                int f2p = f + kw - 1;
                if ((unsigned)f2p >= (unsigned)F2) continue;
                acc += ip[t2 * F2 + f2p] * wp[kh * 3 + kw];
            }
        }
    }
    float v = (acc + bb[c]) * s[c] + sb[c];
    out[idx] = v > 0.f ? v : 0.f;
}

__global__ __launch_bounds__(256) void pool2x_kernel(
    const float* __restrict__ in, float* __restrict__ out, int Fin, int total) {
    int idx = blockIdx.x * 256 + threadIdx.x;
    if (idx >= total) return;
    int Fout = Fin >> 1;
    int f = idx % Fout;
    int rest = idx / Fout;                        // (b*C + c)*T + t
    const float* p = in + (long)rest * Fin + 2 * f;
    float a = p[0], b = p[1];
    out[idx] = a > b ? a : b;
}

// pooled conv3 (B,C2,T,F3) f32 -> x_fc bf16, rows ordered (t*4 + b), cols c*57+f
__global__ __launch_bounds__(256) void pack_x_kernel(
    const float* __restrict__ pooled, unsigned short* __restrict__ xbf) {
    int idx = blockIdx.x * 256 + threadIdx.x;
    const int total = BATCH * T * KFC;
    if (idx >= total) return;
    int col = idx % KFC;
    int row = idx / KFC;
    int t = row >> 2, b = row & 3;
    int c = col / F3, f = col - c * F3;
    float v = pooled[(((b * C2 + c) * T) + t) * F3 + f];
    xbf[idx] = f32_to_bf16(v);
}

// f32 -> bf16 (with row/col zero padding and column offset) for weights
__global__ __launch_bounds__(256) void cvt_bf16_kernel(
    const float* __restrict__ src, unsigned short* __restrict__ dst,
    int srcRows, int dstRows, int srcStride, int colOff, int dstStride, int copyCols) {
    int idx = blockIdx.x * 256 + threadIdx.x;
    int total = dstRows * dstStride;
    if (idx >= total) return;
    int r = idx / dstStride, c = idx - r * dstStride;
    float v = 0.f;
    if (r < srcRows && c < copyCols) v = src[(long)r * srcStride + colOff + c];
    dst[idx] = f32_to_bf16(v);
}

__global__ __launch_bounds__(256) void combine_bias_kernel(
    const float* __restrict__ a, const float* __restrict__ b,
    float* __restrict__ o, int n) {
    int i = blockIdx.x * 256 + threadIdx.x;
    if (i < n) o[i] = a[i] + b[i];
}

// ---------------------------------------------------------------------------
// bf16 WMMA GEMM: C(M,N) f32 = A(M,K) @ B(N,K)^T (+bias), optional bf16 copy.
// One wave per 16x16 C tile. M,N mult of 16; K mult of 32.
// ---------------------------------------------------------------------------
__global__ __launch_bounds__(256) void gemm_bf16_kernel(
    const unsigned short* __restrict__ A, const unsigned short* __restrict__ B,
    const float* __restrict__ bias, float* __restrict__ C,
    unsigned short* __restrict__ Cbf, int M, int N, int K) {
    int lane = threadIdx.x & 31;
    int gwid = (blockIdx.x * blockDim.x + threadIdx.x) >> 5;
    int Nt = N >> 4;
    int tiles = (M >> 4) * Nt;
    if (gwid >= tiles) return;
    int tm = gwid / Nt, tn = gwid - tm * Nt;
    long arow = (long)(tm * 16 + (lane & 15)) * K;
    long brow = (long)(tn * 16 + (lane & 15)) * K;
    v8f acc = {0.f, 0.f, 0.f, 0.f, 0.f, 0.f, 0.f, 0.f};
    for (int k = 0; k < K; k += 32) {
        v16bf a = ld_bf16_op(A, arow, k, lane);
        v16bf b = ld_bf16_op(B, brow, k, lane);
        acc = __builtin_amdgcn_wmma_f32_16x16x32_bf16(false, a, false, b,
                                                      (short)0, acc, false, false);
    }
    int col = tn * 16 + (lane & 15);
    float bv = bias ? bias[col] : 0.f;
    int mbase = tm * 16 + ((lane < 16) ? 0 : 8);
    for (int j = 0; j < 8; ++j) {
        float v = acc[j] + bv;
        long idx = (long)(mbase + j) * N + col;
        C[idx] = v;
        if (Cbf) Cbf[idx] = f32_to_bf16(v);
    }
}

// ---------------------------------------------------------------------------
// Async copy of one 49152-byte P step into LDS, spread over 1024 threads:
// 3 x GLOBAL_LOAD_ASYNC_TO_LDS_B128 per thread (wave ASYNCcnt += 3).
// ---------------------------------------------------------------------------
__device__ __forceinline__ void async_p_prefetch(const float* gsrc,
                                                 unsigned ldsByteOff, int tid) {
#pragma unroll
    for (int i = 0; i < 3; ++i) {
        unsigned lo = ldsByteOff + (unsigned)(tid * 16 + i * 16384);
        unsigned long long ga =
            (unsigned long long)(const char*)gsrc + (unsigned)(tid * 16 + i * 16384);
        asm volatile("global_load_async_to_lds_b128 %0, %1, off"
                     :: "v"(lo), "v"(ga) : "memory");
    }
}

// ---------------------------------------------------------------------------
// Persistent autoregressive LSTM: one workgroup (32 waves), T steps.
// P = precomputed a_t@Wih0[:, :768]^T + bih0 + bhh0, rows ordered t*4+b,
// streamed into an LDS double buffer with the ASYNCcnt split-wait idiom.
// ---------------------------------------------------------------------------
__global__ void __launch_bounds__(1024) lstm_seq_kernel(
    const float* __restrict__ P,
    const unsigned short* __restrict__ Whh0,   // 3072 x 768 bf16
    const unsigned short* __restrict__ Wp,     // 3072 x 192 bf16 (zero-padded)
    const unsigned short* __restrict__ Wih1,   // 3072 x 768 bf16
    const unsigned short* __restrict__ Whh1,   // 3072 x 768 bf16
    const float* __restrict__ bias1,           // bih1 + bhh1
    const unsigned short* __restrict__ Pw,     // 448 x 768 bf16 (zero-padded rows)
    const float* __restrict__ post_b,          // 440
    const float* __restrict__ embed,           // 5 x 2
    float* __restrict__ out) {                 // (B, T, 88)
    extern __shared__ char smem[];
    float* Pbuf         = (float*)smem;                       // 2 x (4 x 3072)
    float* g            = Pbuf + 2 * 4 * G4;                  // 4 x 3072
    float* c0           = g + 4 * G4;                         // 4 x 768
    float* c1           = c0 + 4 * DC;                        // 4 x 768
    float* logits       = c1 + 4 * DC;                        // 4 x 448
    unsigned short* h0  = (unsigned short*)(logits + 4 * NPOSTP); // 16 x 768 bf16
    unsigned short* h1  = h0 + 16 * DC;                       // 16 x 768 bf16
    unsigned short* prv = h1 + 16 * DC;                       // 16 x 192 bf16

    int tid = threadIdx.x;
    int lane = tid & 31, wid = tid >> 5;
    // LDS byte offset of the dynamic segment (Pbuf sits at its start).
    unsigned pbufOff = __builtin_amdgcn_groupstaticsize();

    for (int i = tid; i < 16 * DC; i += 1024) { h0[i] = 0; h1[i] = 0; }
    for (int i = tid; i < 16 * KPREVP; i += 1024) prv[i] = 0;
    for (int i = tid; i < 4 * DC; i += 1024) { c0[i] = 0.f; c1[i] = 0.f; }

    // Prime the pipeline: fetch P[0] into buffer 0.
    async_p_prefetch(P, pbufOff, tid);
    __syncthreads();

    for (int t = 0; t < T; ++t) {
        // Kick prefetch of P[t+1] into the other buffer, then wait for the
        // previous step's 3 async loads (in-order retirement => cnt<=3).
        async_p_prefetch(P + (long)(t + 1) * 4 * G4,
                         pbufOff + (unsigned)(((t + 1) & 1) * PSTEP_BYTES), tid);
        asm volatile("s_wait_asynccnt 3" ::: "memory");
        __syncthreads();                       // make all waves' slices visible
        const float* Pcur = Pbuf + (t & 1) * 4 * G4;

        // ---- phase A: g0 = P[t] + prev@Wp^T + h0@Whh0^T -------------------
        for (int j = 0; j < 6; ++j) {
            int tile = wid * 6 + j;                           // 192 N-tiles
            v8f acc = {0.f, 0.f, 0.f, 0.f, 0.f, 0.f, 0.f, 0.f};
            long brow = (long)(tile * 16 + (lane & 15)) * DC;
            long arow = (long)(lane & 15) * DC;
            for (int k = 0; k < DC; k += 32) {
                v16bf a = ld_bf16_op(h0, arow, k, lane);
                v16bf b = ld_bf16_op(Whh0, brow, k, lane);
                acc = __builtin_amdgcn_wmma_f32_16x16x32_bf16(false, a, false, b,
                                                              (short)0, acc, false, false);
            }
            long browp = (long)(tile * 16 + (lane & 15)) * KPREVP;
            long arowp = (long)(lane & 15) * KPREVP;
            for (int k = 0; k < KPREVP; k += 32) {
                v16bf a = ld_bf16_op(prv, arowp, k, lane);
                v16bf b = ld_bf16_op(Wp, browp, k, lane);
                acc = __builtin_amdgcn_wmma_f32_16x16x32_bf16(false, a, false, b,
                                                              (short)0, acc, false, false);
            }
            if (lane < 16) {
                int col = tile * 16 + lane;
                for (int r = 0; r < 4; ++r)
                    g[r * G4 + col] = acc[r] + Pcur[r * G4 + col];
            }
        }
        __syncthreads();

        // ---- phase B: layer-0 cell activations ----------------------------
        for (int u = tid; u < 4 * DC; u += 1024) {
            int b = u / DC, n = u - b * DC;
            float iv = g[b * G4 + n];
            float fv = g[b * G4 + DC + n];
            float gv = g[b * G4 + 2 * DC + n];
            float ov = g[b * G4 + 3 * DC + n];
            float c = sigf(fv) * c0[u] + sigf(iv) * tanhf(gv);
            c0[u] = c;
            h0[b * DC + n] = f32_to_bf16(sigf(ov) * tanhf(c));
        }
        __syncthreads();

        // ---- phase C: g1 = bias1 + h0@Wih1^T + h1@Whh1^T ------------------
        for (int j = 0; j < 6; ++j) {
            int tile = wid * 6 + j;
            v8f acc = {0.f, 0.f, 0.f, 0.f, 0.f, 0.f, 0.f, 0.f};
            long brow = (long)(tile * 16 + (lane & 15)) * DC;
            long arow = (long)(lane & 15) * DC;
            for (int k = 0; k < DC; k += 32) {
                v16bf a = ld_bf16_op(h0, arow, k, lane);
                v16bf b = ld_bf16_op(Wih1, brow, k, lane);
                acc = __builtin_amdgcn_wmma_f32_16x16x32_bf16(false, a, false, b,
                                                              (short)0, acc, false, false);
            }
            for (int k = 0; k < DC; k += 32) {
                v16bf a = ld_bf16_op(h1, arow, k, lane);
                v16bf b = ld_bf16_op(Whh1, brow, k, lane);
                acc = __builtin_amdgcn_wmma_f32_16x16x32_bf16(false, a, false, b,
                                                              (short)0, acc, false, false);
            }
            if (lane < 16) {
                int col = tile * 16 + lane;
                for (int r = 0; r < 4; ++r)
                    g[r * G4 + col] = acc[r] + bias1[col];
            }
        }
        __syncthreads();

        // ---- phase D: layer-1 cell activations ----------------------------
        for (int u = tid; u < 4 * DC; u += 1024) {
            int b = u / DC, n = u - b * DC;
            float iv = g[b * G4 + n];
            float fv = g[b * G4 + DC + n];
            float gv = g[b * G4 + 2 * DC + n];
            float ov = g[b * G4 + 3 * DC + n];
            float c = sigf(fv) * c1[u] + sigf(iv) * tanhf(gv);
            c1[u] = c;
            h1[b * DC + n] = f32_to_bf16(sigf(ov) * tanhf(c));
        }
        __syncthreads();

        // ---- phase E: logits = h1@post_w^T + post_b (N padded to 448) -----
        if (wid < NPOSTP / 16) {
            int tile = wid;
            v8f acc = {0.f, 0.f, 0.f, 0.f, 0.f, 0.f, 0.f, 0.f};
            long brow = (long)(tile * 16 + (lane & 15)) * DC;
            long arow = (long)(lane & 15) * DC;
            for (int k = 0; k < DC; k += 32) {
                v16bf a = ld_bf16_op(h1, arow, k, lane);
                v16bf b = ld_bf16_op(Pw, brow, k, lane);
                acc = __builtin_amdgcn_wmma_f32_16x16x32_bf16(false, a, false, b,
                                                              (short)0, acc, false, false);
            }
            if (lane < 16) {
                int col = tile * 16 + lane;
                if (col < NPOST)
                    for (int r = 0; r < 4; ++r)
                        logits[r * NPOSTP + col] = acc[r] + post_b[col];
            }
        }
        __syncthreads();

        // ---- phase F: argmax per (b, pitch), write out, feed back embed ---
        if (tid < BATCH * NPITCH) {
            int b = tid / NPITCH, p = tid - b * NPITCH;
            const float* lp = logits + b * NPOSTP + p * NCLS;
            int best = 0;
            float bv = lp[0];
            for (int c = 1; c < NCLS; ++c) {
                float v = lp[c];
                if (v > bv) { bv = v; best = c; }
            }
            out[((long)b * T + t) * NPITCH + p] = (float)best;
            prv[b * KPREVP + 2 * p]     = f32_to_bf16(embed[best * 2 + 0]);
            prv[b * KPREVP + 2 * p + 1] = f32_to_bf16(embed[best * 2 + 1]);
        }
        __syncthreads();
    }
}

// ---------------------------------------------------------------------------
// host launcher
// ---------------------------------------------------------------------------
extern "C" void kernel_launch(void* const* d_in, const int* in_sizes, int n_in,
                              void* d_out, int out_size, void* d_ws, size_t ws_size,
                              hipStream_t stream) {
    (void)in_sizes; (void)n_in; (void)out_size; (void)ws_size;
    const float* conv1_w = (const float*)d_in[0];
    const float* conv1_b = (const float*)d_in[1];
    const float* bn1_s   = (const float*)d_in[2];
    const float* bn1_b   = (const float*)d_in[3];
    const float* conv2_w = (const float*)d_in[4];
    const float* conv2_b = (const float*)d_in[5];
    const float* bn2_s   = (const float*)d_in[6];
    const float* bn2_b   = (const float*)d_in[7];
    const float* conv3_w = (const float*)d_in[8];
    const float* conv3_b = (const float*)d_in[9];
    const float* bn3_s   = (const float*)d_in[10];
    const float* bn3_b   = (const float*)d_in[11];
    const float* fc_w    = (const float*)d_in[12];
    const float* fc_b    = (const float*)d_in[13];
    const float* Wih0    = (const float*)d_in[14];
    const float* Whh0    = (const float*)d_in[15];
    const float* bih0    = (const float*)d_in[16];
    const float* bhh0    = (const float*)d_in[17];
    const float* Wih1    = (const float*)d_in[18];
    const float* Whh1    = (const float*)d_in[19];
    const float* bih1    = (const float*)d_in[20];
    const float* bhh1    = (const float*)d_in[21];
    const float* post_w  = (const float*)d_in[22];
    const float* post_b  = (const float*)d_in[23];
    const float* embed   = (const float*)d_in[24];
    const float* mel     = (const float*)d_in[25];

    char* base = (char*)d_ws;
    size_t off = 0;
    auto alloc = [&](size_t bytes) -> void* {
        void* p = base + off;
        off += (bytes + 255) & ~(size_t)255;
        return p;
    };

    const size_t eConv = (size_t)BATCH * C1 * T * F_IN;          // 90,046,464
    const size_t eP1   = (size_t)BATCH * C1 * T * F2;            // 44,826,624
    const size_t eC3   = (size_t)BATCH * C2 * T * F2;            // 89,653,248
    const size_t eP2   = (size_t)BATCH * C2 * T * F3;            // 44,826,624
    const size_t eX    = (size_t)BATCH * T * KFC;                // 44,826,624
    const size_t eAc   = (size_t)BATCH * T * DC;                 //  6,291,456

    float* R1 = (float*)alloc(eConv * 4);        // conv1 out -> conv3 out -> P
    float* R2 = (float*)alloc(eConv * 4);        // conv2 out -> pooled2
    void*  R3 = alloc(eP1 * 4);                  // pool1 out -> x_fc (bf16)
    float*          acF  = (float*)alloc(eAc * 4);
    unsigned short* acB  = (unsigned short*)alloc(eAc * 2);
    unsigned short* WaccB = (unsigned short*)alloc((size_t)G4 * DC * 2);
    unsigned short* Whh0B = (unsigned short*)alloc((size_t)G4 * DC * 2);
    unsigned short* WpB   = (unsigned short*)alloc((size_t)G4 * KPREVP * 2);
    unsigned short* Wih1B = (unsigned short*)alloc((size_t)G4 * DC * 2);
    unsigned short* Whh1B = (unsigned short*)alloc((size_t)G4 * DC * 2);
    unsigned short* PwB   = (unsigned short*)alloc((size_t)NPOSTP * DC * 2);
    unsigned short* fcwB  = (unsigned short*)alloc((size_t)DC * KFC * 2);
    float* biasP = (float*)alloc((size_t)G4 * 4);
    float* bias1 = (float*)alloc((size_t)G4 * 4);

    float* P = R1;                                // reused after conv stack
    float* pool1 = (float*)R3;
    unsigned short* xbf = (unsigned short*)R3;    // reused after conv3

    auto blocks = [](size_t n) { return (unsigned)((n + 255) / 256); };
    dim3 blk(256);

    // conv frontend
    conv1_kernel<<<blocks(eConv), blk, 0, stream>>>(mel, conv1_w, conv1_b, bn1_s, bn1_b, R1);
    conv2_kernel<<<blocks(eConv), blk, 0, stream>>>(R1, conv2_w, conv2_b, bn2_s, bn2_b, R2);
    pool2x_kernel<<<blocks(eP1), blk, 0, stream>>>(R2, pool1, F_IN, (int)eP1);
    conv3_kernel<<<blocks(eC3), blk, 0, stream>>>(pool1, conv3_w, conv3_b, bn3_s, bn3_b, R1);
    pool2x_kernel<<<blocks(eP2), blk, 0, stream>>>(R1, R2, F2, (int)eP2);
    pack_x_kernel<<<blocks(eX), blk, 0, stream>>>(R2, xbf);

    // bf16 weight copies (+ zero padding)
    cvt_bf16_kernel<<<blocks((size_t)DC * KFC), blk, 0, stream>>>(fc_w, fcwB, DC, DC, KFC, 0, KFC, KFC);
    cvt_bf16_kernel<<<blocks((size_t)G4 * DC), blk, 0, stream>>>(Wih0, WaccB, G4, G4, DIN0, 0, DC, DC);
    cvt_bf16_kernel<<<blocks((size_t)G4 * KPREVP), blk, 0, stream>>>(Wih0, WpB, G4, G4, DIN0, DC, KPREVP, KPREV);
    cvt_bf16_kernel<<<blocks((size_t)G4 * DC), blk, 0, stream>>>(Whh0, Whh0B, G4, G4, DC, 0, DC, DC);
    cvt_bf16_kernel<<<blocks((size_t)G4 * DC), blk, 0, stream>>>(Wih1, Wih1B, G4, G4, DC, 0, DC, DC);
    cvt_bf16_kernel<<<blocks((size_t)G4 * DC), blk, 0, stream>>>(Whh1, Whh1B, G4, G4, DC, 0, DC, DC);
    cvt_bf16_kernel<<<blocks((size_t)NPOSTP * DC), blk, 0, stream>>>(post_w, PwB, NPOST, NPOSTP, DC, 0, DC, DC);
    combine_bias_kernel<<<blocks(G4), blk, 0, stream>>>(bih0, bhh0, biasP, G4);
    combine_bias_kernel<<<blocks(G4), blk, 0, stream>>>(bih1, bhh1, bias1, G4);

    // FC: acoustic = x_fc @ fc_w^T + fc_b   (8192 x 5472 @ 5472 x 768)
    {
        int tiles = (BATCH * T / 16) * (DC / 16);
        gemm_bf16_kernel<<<(tiles * 32 + 255) / 256, blk, 0, stream>>>(
            xbf, fcwB, fc_b, acF, acB, BATCH * T, DC, KFC);
    }
    // Hoisted LSTM-0 input projection: P = acoustic @ Wih0[:, :768]^T + bih0 + bhh0
    {
        int tiles = (BATCH * T / 16) * (G4 / 16);
        gemm_bf16_kernel<<<(tiles * 32 + 255) / 256, blk, 0, stream>>>(
            acB, WaccB, biasP, P, nullptr, BATCH * T, G4, DC);
    }

    // Sequential AR LSTM (persistent single workgroup)
    size_t smem = (size_t)2 * 4 * G4 * 4  // P double buffer (async-to-LDS)
                + (size_t)4 * G4 * 4      // g
                + (size_t)4 * DC * 4 * 2  // c0, c1
                + (size_t)4 * NPOSTP * 4  // logits
                + (size_t)16 * DC * 2 * 2 // h0, h1
                + (size_t)16 * KPREVP * 2;// prev
    lstm_seq_kernel<<<1, 1024, smem, stream>>>(
        P, Whh0B, WpB, Wih1B, Whh1B, bias1, PwB, post_b, embed, (float*)d_out);
}
